// RNNModel_1_layer_30262339568250
// MI455X (gfx1250) — matrix operand order repaired and checked by reading, not compiled, not run on hardware
//
#include <hip/hip_runtime.h>

typedef unsigned short u16;
typedef __attribute__((ext_vector_type(16))) __bf16 v16bf;
typedef __attribute__((ext_vector_type(8)))  float  v8f;

struct F16x2 { uint4 lo, hi; };   // 32 bytes = 16 bf16 values

// ---------- helpers ----------
__device__ __forceinline__ u16 f2bf(float f) {
  unsigned int u = __builtin_bit_cast(unsigned int, f);
  u += 0x7FFFu + ((u >> 16) & 1u);           // round-to-nearest-even
  return (u16)(u >> 16);
}
__device__ __forceinline__ float bf2f(u16 h) {
  unsigned int u = ((unsigned int)h) << 16;
  return __builtin_bit_cast(float, u);
}

// CDNA5 async global->LDS: no data VGPRs, tracked by ASYNCcnt.
// NOTE (CDNA5 ISA): INST_OFFSET is added to BOTH the LDS
// address and the global address, so per-chunk offsets must match on both
// sides (they do: each thread owns 64 contiguous bytes of one row).
#define ASYNC_LD16(ldsaddr, gptr, OFF)                                     \
  asm volatile("global_load_async_to_lds_b128 %0, %1, off offset:" #OFF    \
               :: "v"(ldsaddr), "v"(gptr) : "memory")
#define WAIT_ASYNC0()  asm volatile("s_wait_asynccnt 0x0" ::: "memory")
#define WAIT_XCNT0()   asm volatile("s_wait_xcnt 0x0" ::: "memory")

// ---------- prologue: xw = x @ W_ih^T + b_ih  (f32, [8192,1024]) ----------
__global__ __launch_bounds__(256)
void xw_kernel(const float* __restrict__ x, const float* __restrict__ Wih,
               const float* __restrict__ bih, float* __restrict__ XW) {
  int tid = blockIdx.x * 256 + threadIdx.x;   // 8192*1024 threads
  int b = tid >> 10, h = tid & 1023;
  const float* xr = x   + (size_t)b * 32;
  const float* wr = Wih + (size_t)h * 32;
  float s = bih[h];
  #pragma unroll
  for (int i = 0; i < 32; ++i) s += xr[i] * wr[i];
  XW[tid] = s;
}

// ---------- prologue: pack W_hh into WMMA B-fragment layout (bf16) ----------
// B tile is 32(K) x 16(N). Lane l holds N = 16*nt + (l&15), K = kb*32 + (l>>4)*16 + v,
// v = 0..15 stored contiguously (32 bytes per lane). B[k][n] = W_hh[n][k].
__global__ __launch_bounds__(256)
void pack_kernel(const float* __restrict__ Whh, u16* __restrict__ Wp) {
  int t = blockIdx.x * 256 + threadIdx.x;     // 32*64*32 = 65536 threads
  int kb   = t >> 11;          // K-block of 32 (0..31)
  int nt   = (t >> 5) & 63;    // N-tile of 16 (0..63)
  int lane = t & 31;
  int n     = nt * 16 + (lane & 15);
  int kbase = kb * 32 + (lane >> 4) * 16;
  const float* src = Whh + (size_t)n * 1024 + kbase;
  u16* dst = Wp + ((size_t)t << 4);
  #pragma unroll
  for (int v = 0; v < 16; ++v) dst[v] = f2bf(src[v]);
}

// ---------- prologue: h0 = 0 (bf16) ----------
__global__ __launch_bounds__(256)
void zero_kernel(uint4* __restrict__ p) {
  p[(size_t)blockIdx.x * 256 + threadIdx.x] = make_uint4(0u, 0u, 0u, 0u);
}

// ---------- main recurrent step: Hout = relu(XW + Hin @ W_hh^T + b_hh) ----------
// Grid: (64, 8). Block: 256 threads = 8 waves. WG tile 128(M) x 128(N), K=1024.
// Wave (wr,wc) in 2x4 owns 64x32 => 4x2 accumulators of 16x16 (v8f each).
#define LDA 72   // padded LDS row stride in bf16 units (144B, 16B-aligned)
__global__ __launch_bounds__(256)
void rnn_step_kernel(const u16* __restrict__ Hin, const u16* __restrict__ Wp,
                     const float* __restrict__ XW, const float* __restrict__ bhh,
                     u16* __restrict__ Hout) {
  __shared__ __align__(16) u16 As[2][128 * LDA];

  const int t    = threadIdx.x;
  const int lane = t & 31;
  const int wave = t >> 5;
  const int wr   = wave >> 2;                 // 0..1
  const int wc   = wave & 3;                  // 0..3
  const int mBase  = blockIdx.x * 128;
  const int ntBase = blockIdx.y * 8 + wc * 2; // N-tile-of-16 index base
  const int ln = lane & 15;
  const int lh = lane >> 4;

  v8f acc[4][2];
  #pragma unroll
  for (int m = 0; m < 4; ++m)
    #pragma unroll
    for (int n = 0; n < 2; ++n) acc[m][n] = (v8f)0.0f;

  // cooperative A loads: thread t owns row = t/2, 32 contiguous bf16 (64B)
  // starting at col (t&1)*32 -> 4 async b128 chunks with offsets 0/16/32/48.
  const int arow = t >> 1;
  const int acol = (t & 1) << 5;
  const u16* gp = Hin + (size_t)(mBase + arow) * 1024 + acol;
  const unsigned ldsA[2] = {
    (unsigned)(uintptr_t)&As[0][arow * LDA + acol],
    (unsigned)(uintptr_t)&As[1][arow * LDA + acol],
  };

  // B-fragment byte pointer, strength-reduced: for (kb, nt) the frag lives at
  // Wp_bytes + (kb*64 + nt)*1024 + lane*32; per K-tile (2 kb) that is +131072,
  // per ks +65536, per n +1024, hi-half +16 -> all immediate offsets.
  const char* bpB = (const char*)Wp + (size_t)ntBase * 1024 + (size_t)lane * 32;

  // prime buffer 0 (K-tile 0)
  ASYNC_LD16(ldsA[0], gp, 0);
  ASYNC_LD16(ldsA[0], gp, 16);
  ASYNC_LD16(ldsA[0], gp, 32);
  ASYNC_LD16(ldsA[0], gp, 48);
  WAIT_XCNT0();
  gp += 64;
  WAIT_ASYNC0();
  __syncthreads();

  for (int kt = 0; kt < 16; ++kt) {          // K = 16 x 64
    const int buf = kt & 1;
    if (kt < 15) {                           // async prefetch next tile
      const unsigned ldst = ldsA[buf ^ 1];
      ASYNC_LD16(ldst, gp, 0);
      ASYNC_LD16(ldst, gp, 16);
      ASYNC_LD16(ldst, gp, 32);
      ASYNC_LD16(ldst, gp, 48);
      WAIT_XCNT0();
      gp += 64;
    }
    #pragma unroll
    for (int ks = 0; ks < 2; ++ks) {         // two WMMA-K (32) per LDS tile
      v16bf bfr[2];
      #pragma unroll
      for (int n = 0; n < 2; ++n) {          // B frags: contiguous 32B per lane
        F16x2 f;
        f.lo = *(const uint4*)(bpB + ks * 65536 + n * 1024);
        f.hi = *(const uint4*)(bpB + ks * 65536 + n * 1024 + 16);
        bfr[n] = __builtin_bit_cast(v16bf, f);
      }
      #pragma unroll
      for (int m = 0; m < 4; ++m) {          // A frags from LDS
        const u16* ap = &As[buf][(wr * 64 + m * 16 + ln) * LDA + ks * 32 + lh * 16];
        F16x2 f;
        f.lo = *(const uint4*)ap;            // K = ks*32 + lh*16 + (0..7)
        f.hi = *(const uint4*)(ap + 8);      // K = +8
        v16bf afr = __builtin_bit_cast(v16bf, f);
        #pragma unroll
        for (int n = 0; n < 2; ++n)
          acc[m][n] = __builtin_amdgcn_wmma_f32_16x16x32_bf16(
              false, afr, false, bfr[n], (short)0, acc[m][n], false, false);
      }
    }
    bpB += 131072;                           // next K-tile (2 kb blocks)
    if (kt < 15) WAIT_ASYNC0();              // my prefetch into buf^1 done
    __syncthreads();                         // everyone's prefetch + compute done
  }

  // epilogue: + XW + b_hh, relu, -> bf16
  #pragma unroll
  for (int n = 0; n < 2; ++n) {
    const int col  = (ntBase + n) * 16 + ln;
    const float bias = bhh[col];
    #pragma unroll
    for (int m = 0; m < 4; ++m) {
      const int rbase = mBase + wr * 64 + m * 16 + lh * 8;
      #pragma unroll
      for (int r = 0; r < 8; ++r) {
        const size_t idx = (size_t)(rbase + r) * 1024 + col;
        float v = acc[m][n][r] + XW[idx] + bias;
        v = v > 0.0f ? v : 0.0f;
        Hout[idx] = f2bf(v);
      }
    }
  }
}

// ---------- epilogue head: softmax(h @ fc_W^T + fc_b), O=3 ----------
__global__ __launch_bounds__(256)
void head_kernel(const u16* __restrict__ H, const float* __restrict__ fcW,
                 const float* __restrict__ fcb, float* __restrict__ out) {
  const int wave = threadIdx.x >> 5;
  const int lane = threadIdx.x & 31;
  const int row  = blockIdx.x * 8 + wave;
  const u16* hr = H + (size_t)row * 1024;
  float s0 = 0.f, s1 = 0.f, s2 = 0.f;
  for (int k = lane; k < 1024; k += 32) {
    float hv = bf2f(hr[k]);
    s0 += hv * fcW[k];
    s1 += hv * fcW[1024 + k];
    s2 += hv * fcW[2048 + k];
  }
  #pragma unroll
  for (int off = 16; off > 0; off >>= 1) {
    s0 += __shfl_xor(s0, off, 32);
    s1 += __shfl_xor(s1, off, 32);
    s2 += __shfl_xor(s2, off, 32);
  }
  if (lane == 0) {
    s0 += fcb[0]; s1 += fcb[1]; s2 += fcb[2];
    float mx = fmaxf(s0, fmaxf(s1, s2));
    float e0 = __expf(s0 - mx), e1 = __expf(s1 - mx), e2 = __expf(s2 - mx);
    float inv = 1.0f / (e0 + e1 + e2);
    out[(size_t)row * 3 + 0] = e0 * inv;
    out[(size_t)row * 3 + 1] = e1 * inv;
    out[(size_t)row * 3 + 2] = e2 * inv;
  }
}

extern "C" void kernel_launch(void* const* d_in, const int* in_sizes, int n_in,
                              void* d_out, int out_size, void* d_ws, size_t ws_size,
                              hipStream_t stream) {
  (void)in_sizes; (void)n_in; (void)out_size; (void)ws_size;
  const float* x   = (const float*)d_in[0];
  const float* Wih = (const float*)d_in[1];
  const float* bih = (const float*)d_in[2];
  const float* Whh = (const float*)d_in[3];
  const float* bhh = (const float*)d_in[4];
  const float* fcW = (const float*)d_in[5];
  const float* fcb = (const float*)d_in[6];
  // d_in[7] is length == 128 (problem constant; cannot read device scalar
  // during graph capture, so the loop count is fixed).

  char* ws = (char*)d_ws;
  float* XW = (float*)ws;                                   // 32 MB f32
  u16*   Wp = (u16*)(ws + (size_t)32 * 1024 * 1024);        //  2 MB bf16 packed
  u16*   H0 = (u16*)(ws + (size_t)34 * 1024 * 1024);        // 16 MB bf16
  u16*   H1 = (u16*)(ws + (size_t)50 * 1024 * 1024);        // 16 MB bf16
  u16*   Hb[2] = {H0, H1};

  xw_kernel  <<<(8192 * 1024) / 256, 256, 0, stream>>>(x, Wih, bih, XW);
  pack_kernel<<<65536 / 256,         256, 0, stream>>>(Whh, Wp);
  zero_kernel<<<(8192 * 1024 * 2 / 16) / 256, 256, 0, stream>>>((uint4*)H0);

  for (int i = 0; i < 128; ++i)
    rnn_step_kernel<<<dim3(64, 8), 256, 0, stream>>>(Hb[i & 1], Wp, XW, bhh,
                                                     Hb[(i + 1) & 1]);
  // after 128 steps result is back in H0
  head_kernel<<<1024, 256, 0, stream>>>(H0, fcW, fcb, (float*)d_out);
}
